// ScaledDotProductAttention_3264175145243
// MI455X (gfx1250) — matrix operand order repaired and checked by reading, not compiled, not run on hardware
//
#include <hip/hip_runtime.h>

// Scaled dot-product attention with Gaussian positional bias + bool mask.
// B=64, L=1024, D=64, fp32 in/out. Outputs: output [B,L,D] then attn [B,L,L].
// MI455X (gfx1250): wave32, WMMA f32_16x16x32_f16 (f16 in, f32 accumulate).

#define L_SEQ 1024
#define D_HEAD 64

typedef __attribute__((ext_vector_type(16))) _Float16 v16h;
typedef __attribute__((ext_vector_type(8)))  _Float16 v8h;
typedef __attribute__((ext_vector_type(8)))  float    v8f;

__device__ __forceinline__ v8f wmma16(v16h a, v16h b, v8f c) {
  // D = A(16x32 f16) x B(32x16 f16) + C(16x16 f32)
  return __builtin_amdgcn_wmma_f32_16x16x32_f16(false, a, false, b,
                                                (short)0, c, false, false);
}

__global__ __launch_bounds__(32, 1)
void sdpa_gauss_kernel(const float* __restrict__ Q,
                       const float* __restrict__ K,
                       const float* __restrict__ V,
                       const float* __restrict__ TAU,
                       const unsigned char* __restrict__ MASK,
                       float* __restrict__ OUT,
                       float* __restrict__ ATTN)
{
  // Row pad 1028 (=257*16B) -> conflict-free scatter stores, 16B-aligned rows.
  __shared__ float    S[16][1028];     // scores, then exp-values (64.3 KB)
  __shared__ _Float16 Vt[64][264];     // V^T chunk, f16 (33 KB), pad->bank spread
  __shared__ float    recipL[16];

  const int tid  = threadIdx.x;        // 0..31 (wave32)
  const int half = tid >> 4;           // 0/1
  const int l16  = tid & 15;
  const int b    = blockIdx.x >> 6;    // batch
  const int q0   = (blockIdx.x & 63) << 4;  // 16-row query tile

  const float tau = TAU[0];
  const float sig = tau * tau;
  const float inv_div = 1.0f / (2.0f * sig * sig);   // bias = -(i-j)^2 * inv_div

  // ---------------- Q tile -> A fragments (scale 1/8 folded, exact) --------
  // A layout chosen here (contiguous k; B uses the same): lane l<16: row=l, halves = K[16h..16h+16)
  const int c0 = 16 * half;
  const float* qrow = Q + ((size_t)b * L_SEQ + q0 + l16) * D_HEAD;
  v16h aq[2];
#pragma unroll
  for (int f = 0; f < 2; ++f) {
    const float* p = qrow + 32 * f + c0;
#pragma unroll
    for (int j = 0; j < 8; ++j) {
      aq[f][j]     = (_Float16)(p[j]      * 0.125f);
      aq[f][j + 8] = (_Float16)(p[j + 8] * 0.125f);
    }
  }

  // ---------------- Pass 1: S = (Q/8) K^T  (scores -> LDS) -----------------
  // B layout (32x16): lane holds column n=l16 (a K row); halves = 16
  // contiguous d-dims starting at 16*half  => contiguous reads of K rows.
  for (int kt = 0; kt < 64; ++kt) {
    const float* krow =
        K + ((size_t)b * L_SEQ + (size_t)(kt * 16 + l16)) * D_HEAD + 16 * half;
    v16h b0, b1;
#pragma unroll
    for (int j = 0; j < 16; ++j) {
      b0[j] = (_Float16)krow[j];
      b1[j] = (_Float16)krow[j + 32];
    }
    v8f acc = {};
    acc = wmma16(aq[0], b0, acc);
    acc = wmma16(aq[1], b1, acc);
    // C/D layout: VGPR r holds (row r+8*half, col l16)
#pragma unroll
    for (int r = 0; r < 8; ++r)
      S[r + 8 * half][kt * 16 + l16] = acc[r];
  }

  // ------------- Pass 2: bias + mask + softmax, write attn -----------------
  const float NEG_BIG = -3.0e38f;
  for (int r = 0; r < 16; ++r) {
    const int qi = q0 + r;
    const unsigned int* mrow =
        (const unsigned int*)(MASK + ((size_t)b * L_SEQ + qi) * L_SEQ);
    float* srow = S[r];

    // scan 1: s' = s + bias (or -big if masked); track row max
    float pm = NEG_BIG;
    for (int i = 0; i < 8; ++i) {
      const int c = i * 128 + tid * 4;
      float4 sv = *(float4*)(srow + c);
      const unsigned int mb = mrow[c >> 2];
      float o[4] = {sv.x, sv.y, sv.z, sv.w};
#pragma unroll
      for (int j = 0; j < 4; ++j) {
        const float df = (float)(qi - (c + j));
        float val = o[j] - df * df * inv_div;
        if ((mb >> (8 * j)) & 0xFFu) val = NEG_BIG;
        o[j] = val;
        pm = fmaxf(pm, val);
      }
      sv.x = o[0]; sv.y = o[1]; sv.z = o[2]; sv.w = o[3];
      *(float4*)(srow + c) = sv;
    }
#pragma unroll
    for (int m = 1; m < 32; m <<= 1) pm = fmaxf(pm, __shfl_xor(pm, m, 32));

    // scan 2: e = exp(s'-max); keep e in LDS; row sum
    float ps = 0.0f;
    for (int i = 0; i < 8; ++i) {
      const int c = i * 128 + tid * 4;
      float4 sv = *(float4*)(srow + c);
      sv.x = __expf(sv.x - pm);
      sv.y = __expf(sv.y - pm);
      sv.z = __expf(sv.z - pm);
      sv.w = __expf(sv.w - pm);
      *(float4*)(srow + c) = sv;
      ps += sv.x + sv.y + sv.z + sv.w;
    }
#pragma unroll
    for (int m = 1; m < 32; m <<= 1) ps += __shfl_xor(ps, m, 32);
    const float rl = 1.0f / ps;
    if (tid == 0) recipL[r] = rl;

    // scan 3: attn row = e * (1/sum), coalesced float4 global stores
    float* arow = ATTN + ((size_t)b * L_SEQ + qi) * L_SEQ;
    for (int i = 0; i < 8; ++i) {
      const int c = i * 128 + tid * 4;
      float4 ev = *(float4*)(srow + c);
      ev.x *= rl; ev.y *= rl; ev.z *= rl; ev.w *= rl;
      *(float4*)(arow + c) = ev;
    }
  }

  // ---------------- Pass 3: O = P V  (P = unnormalized e in LDS) -----------
  v8f oacc[4] = {};   // 4 n-tiles of 16 columns each (D=64)
  for (int ch = 0; ch < 4; ++ch) {
    // Stage V[ch*256 .. +256)[0..64) transposed to f16 Vt[d][k]
    const float* vbase = V + ((size_t)b * L_SEQ + ch * 256) * D_HEAD;
    for (int i = 0; i < 128; ++i) {
      const int idx4 = tid + 32 * i;
      const int kk = (idx4 * 4) >> 6;   // 0..255
      const int d  = (idx4 * 4) & 63;   // 0,4,..,60
      const float4 x = *(const float4*)(vbase + (size_t)kk * D_HEAD + d);
      Vt[d + 0][kk] = (_Float16)x.x;
      Vt[d + 1][kk] = (_Float16)x.y;
      Vt[d + 2][kk] = (_Float16)x.z;
      Vt[d + 3][kk] = (_Float16)x.w;
    }
    // Single-wave WG: DS ops are in-order within the wave -> no barrier.
    for (int ki = 0; ki < 8; ++ki) {
      const int kb = ki * 32;
      // A fragment: e rows from S
      const float* ap = &S[l16][ch * 256 + kb + c0];
      v16h af;
#pragma unroll
      for (int j = 0; j < 8; ++j) {
        af[j]     = (_Float16)ap[j];
        af[j + 8] = (_Float16)ap[j + 8];
      }
#pragma unroll
      for (int n = 0; n < 4; ++n) {
        const _Float16* bp = &Vt[n * 16 + l16][kb + 16 * half];
        const v8h lo = *(const v8h*)bp;        // 16B-aligned LDS reads
        const v8h hi = *(const v8h*)(bp + 8);
        v16h bf;
#pragma unroll
        for (int j = 0; j < 8; ++j) { bf[j] = lo[j]; bf[j + 8] = hi[j]; }
        oacc[n] = wmma16(af, bf, oacc[n]);
      }
    }
  }

  // ---------------- Pass 4: scale by 1/rowsum, store output ----------------
  float rl8[8];
#pragma unroll
  for (int r = 0; r < 8; ++r) rl8[r] = recipL[r + 8 * half];
  float* obase = OUT + ((size_t)b * L_SEQ + q0) * D_HEAD;
#pragma unroll
  for (int n = 0; n < 4; ++n)
#pragma unroll
    for (int r = 0; r < 8; ++r)
      obase[(size_t)(r + 8 * half) * D_HEAD + n * 16 + l16] =
          oacc[n][r] * rl8[r];
}

extern "C" void kernel_launch(void* const* d_in, const int* in_sizes, int n_in,
                              void* d_out, int out_size, void* d_ws, size_t ws_size,
                              hipStream_t stream) {
  (void)in_sizes; (void)n_in; (void)out_size; (void)d_ws; (void)ws_size;
  const float*         q    = (const float*)d_in[0];
  const float*         k    = (const float*)d_in[1];
  const float*         v    = (const float*)d_in[2];
  const float*         tau  = (const float*)d_in[3];
  const unsigned char* mask = (const unsigned char*)d_in[4];  // jnp.bool_ = 1 byte

  float* out  = (float*)d_out;                          // [64,1024,64]
  float* attn = out + (size_t)64 * 1024 * 64;           // [64,1024,1024]

  dim3 grid(64 * 64);   // batch * 16-row query tiles
  dim3 block(32);       // one wave32 per workgroup
  sdpa_gauss_kernel<<<grid, block, 0, stream>>>(q, k, v, tau, mask, out, attn);
}
